// ConvFlow_65120294142280
// MI455X (gfx1250) — matrix operand order repaired and checked by reading, not compiled, not run on hardware
//
#include <hip/hip_runtime.h>
#include <hip/hip_bf16.h>
#include <math.h>

typedef __attribute__((ext_vector_type(16))) _Float16 v16h;
typedef __attribute__((ext_vector_type(8)))  _Float16 v8h;
typedef __attribute__((ext_vector_type(8)))  float    v8f;

#define C_CH     192
#define T_SEQ    4096
#define NBINS    10
#define TB       5.0f
#define MIN_BIN  0.001f
#define MIN_DER  0.001f
#define TOUT     64          // output timesteps per workgroup
#define HALO     13          // 1 + 3 + 9 receptive-field halo
#define TW       (TOUT + 2*HALO)   // 90
#define TROWS    96          // padded time rows for act/z tail tiles
#define TS       96          // h time stride (floats): lane-contiguous t
#define CS       200         // act/z channel stride (halfs): 100 dwords -> conflict-free
#define WS       200         // weight k stride (halfs)
#define NTHREADS 256
#define NWAVES   8

__device__ __forceinline__ float gelu_exact(float x) {
    return 0.5f * x * (1.0f + erff(x * 0.7071067811865475f));
}

__global__ void zero_logdet_kernel(float* p, int n) {
    int i = blockIdx.x * blockDim.x + threadIdx.x;
    if (i < n) p[i] = 0.0f;
}

__global__ __launch_bounds__(NTHREADS, 1)
void convflow_fused_kernel(const float* __restrict__ x,
                           const float* __restrict__ xmask,
                           const float* __restrict__ pre_w,  const float* __restrict__ pre_b,
                           const float* __restrict__ sep_w,  const float* __restrict__ sep_b,
                           const float* __restrict__ pw_w,   const float* __restrict__ pw_b,
                           const float* __restrict__ ln1_g,  const float* __restrict__ ln1_b,
                           const float* __restrict__ ln2_g,  const float* __restrict__ ln2_b,
                           const float* __restrict__ proj_w, const float* __restrict__ proj_b,
                           float* __restrict__ out, float* __restrict__ logdet)
{
    __shared__ __align__(16) float    hbuf[C_CH * TS];    // 73.7 KB, h[c][t] f32
    __shared__ __align__(16) _Float16 wbuf[C_CH * WS];    // 76.8 KB, W[cout][cin] f16
    __shared__ __align__(16) _Float16 abuf[TROWS * CS];   // 38.4 KB, act[t][c] f16
    __shared__ __align__(16) _Float16 zbuf[TROWS * CS];   // 38.4 KB, z[t][c] f16
    __shared__ float mbuf[TROWS];
    __shared__ float xbuf[TROWS];
    __shared__ float p_sw0[C_CH], p_sw1[C_CH], p_sw2[C_CH], p_sb[C_CH];
    __shared__ float p_g1[C_CH], p_b1[C_CH], p_g2[C_CH], p_b2[C_CH], p_pwb[C_CH];
    __shared__ float zp[(3*NBINS - 1) * TOUT];            // 29x64 projection output
    __shared__ float s_lad[TOUT];

    const int tid = threadIdx.x;
    const int b   = blockIdx.y;
    const int t0  = blockIdx.x * TOUT;

    // ---- stage x0 tile + mask (halo; zero outside sequence == zero-padding) ----
    for (int i = tid; i < TW; i += NTHREADS) {
        int gt = t0 - HALO + i;
        bool inr = (gt >= 0) && (gt < T_SEQ);
        xbuf[i] = inr ? x[(size_t)b * 2 * T_SEQ + gt] : 0.0f;
        mbuf[i] = inr ? xmask[(size_t)b * T_SEQ + gt] : 0.0f;
    }
    __syncthreads();

    // ---- pre 1x1 conv: h[c][t] = pre_w[c]*x0[t] + pre_b[c] ----
    for (int i = tid; i < C_CH * TW; i += NTHREADS) {
        int c = i / TW, t = i - c * TW;
        hbuf[c * TS + t] = pre_w[c] * xbuf[t] + pre_b[c];
    }

    int lo = 0, hi = TW, dil = 1;
    for (int layer = 0; layer < 3; ++layer) {
        __syncthreads();
        // ---- stage per-layer params + pointwise weights (f32 -> f16) ----
        for (int c = tid; c < C_CH; c += NTHREADS) {
            p_sw0[c] = sep_w[(layer * C_CH + c) * 3 + 0];
            p_sw1[c] = sep_w[(layer * C_CH + c) * 3 + 1];
            p_sw2[c] = sep_w[(layer * C_CH + c) * 3 + 2];
            p_sb[c]  = sep_b[layer * C_CH + c];
            p_g1[c]  = ln1_g[layer * C_CH + c];
            p_b1[c]  = ln1_b[layer * C_CH + c];
            p_g2[c]  = ln2_g[layer * C_CH + c];
            p_b2[c]  = ln2_b[layer * C_CH + c];
            p_pwb[c] = pw_b[layer * C_CH + c];
        }
        const float* wl = pw_w + (size_t)layer * C_CH * C_CH;
        for (int i = tid; i < C_CH * C_CH; i += NTHREADS) {
            int co = i / C_CH, ci = i - co * C_CH;
            wbuf[co * WS + ci] = (_Float16)wl[i];
        }
        __syncthreads();

        const int nlo = lo + dil, nhi = hi - dil;
        const int W = nhi - nlo;

        // ---- depthwise dilated conv + LN1 + GELU -> act[t][c] f16 ----
        for (int t = nlo + tid; t < nhi; t += NTHREADS) {
            float mL = mbuf[t - dil], mC = mbuf[t], mR = mbuf[t + dil];
            float s = 0.f, ss = 0.f;
            for (int c = 0; c < C_CH; ++c) {
                float v = p_sw0[c] * hbuf[c * TS + t - dil] * mL
                        + p_sw1[c] * hbuf[c * TS + t      ] * mC
                        + p_sw2[c] * hbuf[c * TS + t + dil] * mR + p_sb[c];
                s += v; ss += v * v;
            }
            float mean = s * (1.0f / C_CH);
            float var  = ss * (1.0f / C_CH) - mean * mean;
            float rs   = rsqrtf(var + 1e-5f);
            for (int c = 0; c < C_CH; ++c) {
                float v = p_sw0[c] * hbuf[c * TS + t - dil] * mL
                        + p_sw1[c] * hbuf[c * TS + t      ] * mC
                        + p_sw2[c] * hbuf[c * TS + t + dil] * mR + p_sb[c];
                float u = p_g1[c] * (v - mean) * rs + p_b1[c];
                abuf[t * CS + c] = (_Float16)gelu_exact(u);
            }
        }
        __syncthreads();

        // ---- pointwise 192x192 conv as WMMA GEMM: z = W * act ----
        // Schedule: task = (n-tile, m-group of 3). B operand (activations) is
        // loaded once into 48 VGPRs and reused across 3 m-tiles x 6 k-steps,
        // cutting LDS traffic to ~2.67 ds_load_b128 per WMMA. Task counts
        // (ntl*4 = 24 or 16) divide evenly across the 8 waves.
        {
            const int lane = tid & 31;
            const int wave = tid >> 5;
            const int half = lane >> 4;
            const int lan  = lane & 15;
            const int ntl  = (W + 15) >> 4;
            const int ntasks = ntl * 4;
            for (int task = wave; task < ntasks; task += NWAVES) {
                int nt = task >> 2;
                int mg = task & 3;                     // m-tiles [mg*3, mg*3+3)
                int trow = nlo + nt * 16 + lan;        // N index -> time
                int tr = trow < TROWS ? trow : (TROWS - 1);
                // preload B (32x16 f16) for all 6 k-tiles
                v16h bv[6];
                #pragma unroll
                for (int kt = 0; kt < 6; ++kt) {
                    const _Float16* bp = &abuf[tr * CS + kt * 32 + half * 16];
                    v8h b_lo = *(const v8h*)(bp);
                    v8h b_hi = *(const v8h*)(bp + 8);
                    bv[kt] = __builtin_shufflevector(b_lo, b_hi,
                        0,1,2,3,4,5,6,7,8,9,10,11,12,13,14,15);
                }
                #pragma unroll
                for (int mi = 0; mi < 3; ++mi) {
                    int mt = mg * 3 + mi;
                    v8f acc = {};
                    #pragma unroll
                    for (int kt = 0; kt < 6; ++kt) {
                        // A (16x32 f16): lanes 0-15 M rows; element s -> K=half*16+s
                        const _Float16* ap =
                            &wbuf[(mt * 16 + lan) * WS + kt * 32 + half * 16];
                        v8h a_lo = *(const v8h*)(ap);
                        v8h a_hi = *(const v8h*)(ap + 8);
                        v16h av = __builtin_shufflevector(a_lo, a_hi,
                            0,1,2,3,4,5,6,7,8,9,10,11,12,13,14,15);
                        acc = __builtin_amdgcn_wmma_f32_16x16x32_f16(
                            false, av, false, bv[kt], (short)0, acc, false, false);
                    }
                    if (trow < nhi) {                  // D layout: M = r + 8*half, N = lan
                        #pragma unroll
                        for (int r = 0; r < 8; ++r) {
                            int co = mt * 16 + r + 8 * half;
                            zbuf[trow * CS + co] = (_Float16)(acc[r] + p_pwb[co]);
                        }
                    }
                }
            }
        }
        __syncthreads();

        // ---- LN2 + GELU + residual into h ----
        for (int t = nlo + tid; t < nhi; t += NTHREADS) {
            float s = 0.f, ss = 0.f;
            for (int c = 0; c < C_CH; ++c) {
                float v = (float)zbuf[t * CS + c];
                s += v; ss += v * v;
            }
            float mean = s * (1.0f / C_CH);
            float var  = ss * (1.0f / C_CH) - mean * mean;
            float rs   = rsqrtf(var + 1e-5f);
            for (int c = 0; c < C_CH; ++c) {
                float v = (float)zbuf[t * CS + c];
                float u = p_g2[c] * (v - mean) * rs + p_b2[c];
                hbuf[c * TS + t] += gelu_exact(u);
            }
        }
        lo = nlo; hi = nhi; dil *= 3;
    }
    __syncthreads();

    // ---- projection 192 -> 29 (h*mask, +bias, *mask) ----
    for (int i = tid; i < (3 * NBINS - 1) * TOUT; i += NTHREADS) {
        int j = i / TOUT, t = i - j * TOUT;
        int tt = HALO + t;
        float m = mbuf[tt];
        float s = proj_b[j];
        const float* pw = proj_w + (size_t)j * C_CH;
        for (int c = 0; c < C_CH; ++c)
            s += pw[c] * hbuf[c * TS + tt] * m;
        zp[j * TOUT + t] = s * m;
    }
    __syncthreads();

    // ---- rational-quadratic spline per timestep ----
    for (int t = tid; t < TOUT; t += NTHREADS) {
        const float scale = 13.856406460551018f; // sqrt(192)
        float uw[NBINS], uh[NBINS];
        for (int k = 0; k < NBINS; ++k) {
            uw[k] = zp[k * TOUT + t] / scale;
            uh[k] = zp[(NBINS + k) * TOUT + t] / scale;
        }
        float der[NBINS + 1];
        der[0] = 1.0f; der[NBINS] = 1.0f;   // softplus(const) == 1 - MIN_DER exactly
        for (int k = 1; k < NBINS; ++k) {
            float u = zp[(2 * NBINS + k - 1) * TOUT + t];
            float sp = (u > 20.f) ? u : log1pf(expf(u));
            der[k] = MIN_DER + sp;
        }
        float mw = uw[0], mh = uh[0];
        for (int k = 1; k < NBINS; ++k) { mw = fmaxf(mw, uw[k]); mh = fmaxf(mh, uh[k]); }
        float sw = 0.f, sh = 0.f, ew[NBINS], eh[NBINS];
        for (int k = 0; k < NBINS; ++k) {
            ew[k] = expf(uw[k] - mw); sw += ew[k];
            eh[k] = expf(uh[k] - mh); sh += eh[k];
        }
        float cumw[NBINS + 1], cumh[NBINS + 1];
        cumw[0] = -TB; cumh[0] = -TB;
        float cw = 0.f, ch = 0.f;
        for (int k = 0; k < NBINS; ++k) {
            cw += MIN_BIN + (1.0f - MIN_BIN * NBINS) * (ew[k] / sw);
            ch += MIN_BIN + (1.0f - MIN_BIN * NBINS) * (eh[k] / sh);
            cumw[k + 1] = 2.0f * TB * cw - TB;
            cumh[k + 1] = 2.0f * TB * ch - TB;
        }
        cumw[NBINS] = TB; cumh[NBINS] = TB;

        float x1v = x[(size_t)b * 2 * T_SEQ + T_SEQ + (t0 + t)];
        bool inside = (x1v >= -TB) && (x1v <= TB);
        float xin = fminf(fmaxf(x1v, -TB), TB);
        int cnt = 0;
        for (int j = 0; j <= NBINS; ++j) {
            float loc = cumw[j] + ((j == NBINS) ? 1e-6f : 0.0f);
            cnt += (xin >= loc) ? 1 : 0;
        }
        int bin = cnt - 1; bin = bin < 0 ? 0 : (bin > NBINS - 1 ? NBINS - 1 : bin);
        float icw = cumw[bin], iw = cumw[bin + 1] - cumw[bin];
        float ich = cumh[bin], ih = cumh[bin + 1] - cumh[bin];
        float dl = der[bin], dr = der[bin + 1];
        float delta = ih / iw;
        float th  = (xin - icw) / iw;
        float t1m = th * (1.0f - th);
        float num = ih * (delta * th * th + dl * t1m);
        float den = delta + (dl + dr - 2.0f * delta) * t1m;
        float yv  = ich + num / den;
        float dnm = delta * delta * (dr * th * th + 2.0f * delta * t1m
                                     + dl * (1.0f - th) * (1.0f - th));
        float lad = logf(dnm) - 2.0f * logf(den);
        float o   = inside ? yv : x1v;
        lad = inside ? lad : 0.0f;

        float m = mbuf[HALO + t];
        out[(size_t)b * 2 * T_SEQ + (t0 + t)]         = xbuf[HALO + t] * m;
        out[(size_t)b * 2 * T_SEQ + T_SEQ + (t0 + t)] = o * m;
        s_lad[t] = lad * m;
    }
    __syncthreads();
    if (tid == 0) {
        float s = 0.f;
        for (int i = 0; i < TOUT; ++i) s += s_lad[i];
        atomicAdd(&logdet[b], s);
    }
}

extern "C" void kernel_launch(void* const* d_in, const int* in_sizes, int n_in,
                              void* d_out, int out_size, void* d_ws, size_t ws_size,
                              hipStream_t stream) {
    const float* x      = (const float*)d_in[0];
    const float* xmask  = (const float*)d_in[1];
    const float* pre_w  = (const float*)d_in[2];
    const float* pre_b  = (const float*)d_in[3];
    const float* sep_w  = (const float*)d_in[4];
    const float* sep_b  = (const float*)d_in[5];
    const float* pw_w   = (const float*)d_in[6];
    const float* pw_b   = (const float*)d_in[7];
    const float* ln1_g  = (const float*)d_in[8];
    const float* ln1_b  = (const float*)d_in[9];
    const float* ln2_g  = (const float*)d_in[10];
    const float* ln2_b  = (const float*)d_in[11];
    const float* proj_w = (const float*)d_in[12];
    const float* proj_b = (const float*)d_in[13];

    const int Bn = in_sizes[0] / (2 * T_SEQ);
    float* out    = (float*)d_out;
    float* logdet = out + (size_t)Bn * 2 * T_SEQ;

    zero_logdet_kernel<<<dim3((Bn + 63) / 64), dim3(64), 0, stream>>>(logdet, Bn);

    dim3 grid(T_SEQ / TOUT, Bn);
    convflow_fused_kernel<<<grid, dim3(NTHREADS), 0, stream>>>(
        x, xmask, pre_w, pre_b, sep_w, sep_b, pw_w, pw_b,
        ln1_g, ln1_b, ln2_g, ln2_b, proj_w, proj_b, out, logdet);
}
